// _DeformAttnLayer_35330400977532
// MI455X (gfx1250) — matrix operand order repaired and checked by reading, not compiled, not run on hardware
//
#include <hip/hip_runtime.h>
#include <math.h>

// Problem constants (from reference)
#define BB    8
#define LQ    10000
#define CC    256
#define NHH   8
#define NPP   4
#define HDD   32
#define HS    100
#define WS_   100
#define LIN   10000
#define MROWS (BB * LQ)   // 80000 rows, divisible by 64

// GEMM tiling
#define BM 64
#define BN 128
#define BK 32

typedef __attribute__((ext_vector_type(16))) __bf16 v16bf;
typedef __attribute__((ext_vector_type(8)))  float  v8f;

__device__ __forceinline__ unsigned short f2bf(float f) {
    unsigned u = __float_as_uint(f);
    u += 0x7FFFu + ((u >> 16) & 1u);     // round-to-nearest-even
    return (unsigned short)(u >> 16);
}
__device__ __forceinline__ float bf2f(unsigned short h) {
    return __uint_as_float(((unsigned)h) << 16);
}

union FragU {
    v16bf v;
    uint4 q[2];
};

// Generic->LDS offset: flat addresses of LDS carry the byte offset in addr[31:0].
__device__ __forceinline__ unsigned ldsoff(const void* p) {
    return (unsigned)(uintptr_t)p;
}
// CDNA5 async copy global -> LDS (16B per lane), tracked by ASYNCcnt.
__device__ __forceinline__ void async_b128(const void* lds, const void* g) {
    asm volatile("global_load_async_to_lds_b128 %0, %1, off"
                 :: "v"(ldsoff(lds)), "v"(g) : "memory");
}
#define WAIT_ASYNC(n) asm volatile("s_wait_asynccnt %0" :: "n"(n) : "memory")

// Weight convert + transpose: W fp32 [K][N]  ->  Wt bf16 [N][K]
__global__ __launch_bounds__(256)
void cvt_w(const float* __restrict__ W, unsigned short* __restrict__ Wt, int K, int N) {
    int idx = blockIdx.x * 256 + threadIdx.x;
    if (idx < K * N) {
        int k = idx / N, n = idx - k * N;
        Wt[(size_t)n * K + k] = f2bf(W[idx]);
    }
}

// Tiled WMMA GEMM:  out[M][N] = A[M][K] * Wt[N][K]^T + bias
// 128 threads = 4 waves, workgroup tile 64x128; each wave 32x64 (2x4 WMMA tiles).
// Double-buffered LDS fed by GLOBAL_LOAD_ASYNC_TO_LDS_B128, pipelined one K-chunk ahead.
template <int ABF16>
__global__ __launch_bounds__(128)
void gemm_ker(const void* __restrict__ Ain,
              const unsigned short* __restrict__ Wt,
              const float* __restrict__ bias,
              float* __restrict__ outF, unsigned short* __restrict__ outB,
              int M, int N, int K, int relu)
{
    __shared__ __attribute__((aligned(16))) unsigned short sB[2][BN][40];
    __shared__ __attribute__((aligned(16))) unsigned short sA[ABF16 ? 2 : 1][BM][40];

    const int tid  = threadIdx.x;
    const int m0   = blockIdx.y * BM;
    const int n0   = blockIdx.x * BN;
    const int wave = tid >> 5, lane = tid & 31;
    const int wm   = wave >> 1, wn = wave & 1;   // waves: 2 in M x 2 in N
    const int lm   = lane & 15, lh = lane >> 4;

    // ---- async issue helpers (uniform issue count: OOB rows clamped, not skipped;
    //      garbage lands only in accumulator columns the epilogue never stores) ----
    auto issueB = [&](int buf, int kk) {
        #pragma unroll
        for (int j = 0; j < 4; ++j) {               // 128x32 bf16 tile: 4 x b128 per thread
            int i8 = tid + 128 * j;
            int r = i8 >> 2, c = (i8 & 3) << 3;
            int n = n0 + r; n = n < N ? n : N - 1;
            async_b128(&sB[buf][r][c], Wt + (size_t)n * K + kk + c);
        }
    };
    auto issueA = [&](int buf, int kk) {            // bf16 A: 64x32 tile, 2 x b128
        const unsigned short* A = (const unsigned short*)Ain;
        #pragma unroll
        for (int j = 0; j < 2; ++j) {
            int i8 = tid + 128 * j;
            int r = i8 >> 2, c = (i8 & 3) << 3;
            async_b128(&sA[buf][r][c], A + (size_t)(m0 + r) * K + kk + c);
        }
    };
    auto stageA_f32 = [&](int kk) {                 // fp32 A: load+convert+ds_store
        const float* A = (const float*)Ain;
        #pragma unroll
        for (int j = 0; j < 4; ++j) {
            int i4 = tid + 128 * j;
            int r = i4 >> 3, c = (i4 & 7) << 2;
            float4 v = *reinterpret_cast<const float4*>(A + (size_t)(m0 + r) * K + kk + c);
            uint2 p;
            p.x = (unsigned)f2bf(v.x) | ((unsigned)f2bf(v.y) << 16);
            p.y = (unsigned)f2bf(v.z) | ((unsigned)f2bf(v.w) << 16);
            *reinterpret_cast<uint2*>(&sA[0][r][c]) = p;
        }
    };

    v8f zero = {0.f, 0.f, 0.f, 0.f, 0.f, 0.f, 0.f, 0.f};
    v8f acc[2][4];
    #pragma unroll
    for (int i = 0; i < 2; ++i)
        #pragma unroll
        for (int j = 0; j < 4; ++j) acc[i][j] = zero;

    // prologue: chunk 0 in flight into buffer 0
    if constexpr (ABF16 != 0) issueA(0, 0);
    issueB(0, 0);

    const int nchunk = K / BK;
    for (int ci = 0; ci < nchunk; ++ci) {
        const int buf = ci & 1;
        const int k0  = ci * BK;

        if constexpr (ABF16 == 0) {
            if (ci + 1 < nchunk) {      // speculative prefetch of next A tile
                const char* pn = (const char*)Ain +
                    4 * ((size_t)(m0 + (tid & 63)) * K + k0 + BK);
                __builtin_prefetch(pn, 0, 3);
            }
            stageA_f32(k0);
        }

        if (ci + 1 < nchunk) {          // pipeline next chunk into the other buffer
            if constexpr (ABF16 != 0) {
                issueA(buf ^ 1, k0 + BK);
                issueB(buf ^ 1, k0 + BK);
                WAIT_ASYNC(6);          // 6 outstanding = next chunk; current done
            } else {
                issueB(buf ^ 1, k0 + BK);
                WAIT_ASYNC(4);
            }
        } else {
            WAIT_ASYNC(0);
        }
        __syncthreads();

        // ---- per-lane fragments (ISA 7.12.2 layouts) ----
        const int ab = ABF16 ? buf : 0;
        FragU fa[2], fb[4];
        #pragma unroll
        for (int i = 0; i < 2; ++i) {
            int r = wm * 32 + i * 16 + lm;                 // A: M = lane&15
            fa[i].q[0] = *reinterpret_cast<const uint4*>(&sA[ab][r][lh * 8]);
            fa[i].q[1] = *reinterpret_cast<const uint4*>(&sA[ab][r][16 + lh * 8]);
        }
        #pragma unroll
        for (int j = 0; j < 4; ++j) {
            int r = wn * 64 + j * 16 + lm;                 // B: N = lane&15
            fb[j].q[0] = *reinterpret_cast<const uint4*>(&sB[buf][r][lh * 8]);
            fb[j].q[1] = *reinterpret_cast<const uint4*>(&sB[buf][r][16 + lh * 8]);
        }
        #pragma unroll
        for (int i = 0; i < 2; ++i)
            #pragma unroll
            for (int j = 0; j < 4; ++j)
                acc[i][j] = __builtin_amdgcn_wmma_f32_16x16x32_bf16(
                    false, fa[i].v, false, fb[j].v, (short)0, acc[i][j], false, false);
        __syncthreads();   // protect buffers before next iteration overwrites
    }

    // ---- epilogue: bias (+relu), fp32 and/or bf16 stores ----
    #pragma unroll
    for (int j = 0; j < 4; ++j) {
        int col = n0 + wn * 64 + j * 16 + lm;
        if (col >= N) continue;
        float bv = bias ? bias[col] : 0.f;
        #pragma unroll
        for (int i = 0; i < 2; ++i) {
            int row0 = m0 + wm * 32 + i * 16 + lh * 8;     // C: M = r + 8*(lane>>4)
            #pragma unroll
            for (int r = 0; r < 8; ++r) {
                float v = acc[i][j][r] + bv;
                if (relu) v = v > 0.f ? v : 0.f;
                size_t off = (size_t)(row0 + r) * N + col;
                if (outF) outF[off] = v;
                if (outB) outB[off] = f2bf(v);
            }
        }
    }
}

// Deformable sampling: one wave per (b, q, head); lane = channel d (HD==32==wave32).
__global__ __launch_bounds__(256)
void msda_sample(const float* __restrict__ off, const float* __restrict__ aw,
                 const float* __restrict__ refp,
                 const unsigned short* __restrict__ value,
                 unsigned short* __restrict__ attn)
{
    int flat = blockIdx.x * 8 + (threadIdx.x >> 5);
    if (flat >= BB * LQ * NHH) return;
    int lane = threadIdx.x & 31;
    int h  = flat % NHH;
    int rq = flat / NHH;          // b*LQ + q
    int b  = rq / LQ;

    float rx = refp[rq * 2 + 0];
    float ry = refp[rq * 2 + 1];
    const float* op = off + (size_t)rq * (NHH * NPP * 2) + h * (NPP * 2);
    const float* ap = aw  + (size_t)rq * (NHH * NPP)     + h * NPP;

    float a0 = ap[0], a1 = ap[1], a2 = ap[2], a3 = ap[3];
    float mx = fmaxf(fmaxf(a0, a1), fmaxf(a2, a3));
    float e0 = __expf(a0 - mx), e1 = __expf(a1 - mx);
    float e2 = __expf(a2 - mx), e3 = __expf(a3 - mx);
    float inv = 1.f / (e0 + e1 + e2 + e3);
    float w4[4] = {e0 * inv, e1 * inv, e2 * inv, e3 * inv};

    const unsigned short* vb = value + (size_t)b * LIN * CC + h * HDD + lane;
    float acc = 0.f;
    #pragma unroll
    for (int p = 0; p < NPP; ++p) {
        // pixel coord = (ref + off/W)*W - 0.5 == ref*W + off - 0.5
        float x = rx * (float)WS_ + op[p * 2 + 0] - 0.5f;
        float y = ry * (float)HS  + op[p * 2 + 1] - 0.5f;
        float xf = floorf(x), yf = floorf(y);
        float lx = x - xf, ly = y - yf;
        int x0 = (int)xf, y0 = (int)yf;
        float wp = w4[p];
        float cw[4] = {(1.f - lx) * (1.f - ly), lx * (1.f - ly),
                       (1.f - lx) * ly,         lx * ly};
        int xs[4] = {x0, x0 + 1, x0, x0 + 1};
        int ys[4] = {y0, y0, y0 + 1, y0 + 1};
        #pragma unroll
        for (int c2 = 0; c2 < 4; ++c2) {
            int xi = xs[c2], yi = ys[c2];
            if (xi >= 0 && xi < WS_ && yi >= 0 && yi < HS) {
                float v = bf2f(vb[(size_t)(yi * WS_ + xi) * CC]);
                acc += wp * cw[c2] * v;
            }
        }
    }
    attn[(size_t)rq * CC + h * HDD + lane] = f2bf(acc);
}

// Fused residual + LayerNorm; one wave per row of 256, wave32 shuffle reduce.
__global__ __launch_bounds__(256)
void add_ln(const float* __restrict__ x, const float* __restrict__ y,
            const float* __restrict__ g, const float* __restrict__ be,
            float* __restrict__ outF, unsigned short* __restrict__ outB, int M)
{
    int row = blockIdx.x * 8 + (threadIdx.x >> 5);
    if (row >= M) return;
    int lane = threadIdx.x & 31;
    float v[8];
    float s = 0.f;
    #pragma unroll
    for (int k = 0; k < 8; ++k) {
        int c = lane + 32 * k;
        v[k] = x[(size_t)row * CC + c] + y[(size_t)row * CC + c];
        s += v[k];
    }
    #pragma unroll
    for (int d = 16; d >= 1; d >>= 1) s += __shfl_xor(s, d, 32);
    float mean = s * (1.f / CC);
    float q = 0.f;
    #pragma unroll
    for (int k = 0; k < 8; ++k) { float t = v[k] - mean; q += t * t; }
    #pragma unroll
    for (int d = 16; d >= 1; d >>= 1) q += __shfl_xor(q, d, 32);
    float rstd = rsqrtf(q * (1.f / CC) + 1e-5f);
    #pragma unroll
    for (int k = 0; k < 8; ++k) {
        int c = lane + 32 * k;
        float o = (v[k] - mean) * rstd * g[c] + be[c];
        if (outF) outF[(size_t)row * CC + c] = o;
        if (outB) outB[(size_t)row * CC + c] = f2bf(o);
    }
}

extern "C" void kernel_launch(void* const* d_in, const int* in_sizes, int n_in,
                              void* d_out, int out_size, void* d_ws, size_t ws_size,
                              hipStream_t stream)
{
    (void)in_sizes; (void)n_in; (void)out_size; (void)ws_size;
    const float* query = (const float*)d_in[0];
    const float* src   = (const float*)d_in[1];
    const float* refp  = (const float*)d_in[2];
    // d_in[3] spatial_shapes / d_in[4] level_start_index: compile-time constants (100,100)
    const float* W_off = (const float*)d_in[5];
    const float* b_off = (const float*)d_in[6];
    const float* W_aw  = (const float*)d_in[7];
    const float* b_aw  = (const float*)d_in[8];
    const float* W_val = (const float*)d_in[9];
    const float* b_val = (const float*)d_in[10];
    const float* W_out = (const float*)d_in[11];
    const float* b_out = (const float*)d_in[12];
    const float* g1    = (const float*)d_in[13];
    const float* be1   = (const float*)d_in[14];
    const float* W1    = (const float*)d_in[15];
    const float* b1    = (const float*)d_in[16];
    const float* W2    = (const float*)d_in[17];
    const float* b2    = (const float*)d_in[18];
    const float* g2    = (const float*)d_in[19];
    const float* be2   = (const float*)d_in[20];
    float* out = (float*)d_out;

    char* ws = (char*)d_ws;
    size_t o = 0;
    auto alloc = [&](size_t bytes) {
        char* p = ws + o;
        o += (bytes + 255) & ~(size_t)255;
        return p;
    };

    unsigned short* WvT   = (unsigned short*)alloc((size_t)256 * 256 * 2);
    unsigned short* WoffT = (unsigned short*)alloc((size_t)64  * 256 * 2);
    unsigned short* WawT  = (unsigned short*)alloc((size_t)32  * 256 * 2);
    unsigned short* WoutT = (unsigned short*)alloc((size_t)256 * 256 * 2);
    unsigned short* W1T   = (unsigned short*)alloc((size_t)1024 * 256 * 2);
    unsigned short* W2T   = (unsigned short*)alloc((size_t)256 * 1024 * 2);
    unsigned short* valB  = (unsigned short*)alloc((size_t)MROWS * 256 * 2);
    float*          offF  = (float*)alloc((size_t)MROWS * 64 * 4);
    float*          awF   = (float*)alloc((size_t)MROWS * 32 * 4);
    unsigned short* attnB = (unsigned short*)alloc((size_t)MROWS * 256 * 2);
    float*          scrF  = (float*)alloc((size_t)MROWS * 256 * 4);  // attn_out, reused as ffn_out
    float*          qF    = (float*)alloc((size_t)MROWS * 256 * 4);
    unsigned short* qB    = (unsigned short*)alloc((size_t)MROWS * 256 * 2);
    unsigned short* h1    = (unsigned short*)alloc((size_t)MROWS * 1024 * 2);

    // 1) weights -> bf16, transposed to [N][K]
    cvt_w<<<(256 * 256 + 255) / 256, 256, 0, stream>>>(W_val, WvT, 256, 256);
    cvt_w<<<(256 * 64  + 255) / 256, 256, 0, stream>>>(W_off, WoffT, 256, 64);
    cvt_w<<<(256 * 32  + 255) / 256, 256, 0, stream>>>(W_aw,  WawT, 256, 32);
    cvt_w<<<(256 * 256 + 255) / 256, 256, 0, stream>>>(W_out, WoutT, 256, 256);
    cvt_w<<<(256 * 1024 + 255) / 256, 256, 0, stream>>>(W1, W1T, 256, 1024);
    cvt_w<<<(1024 * 256 + 255) / 256, 256, 0, stream>>>(W2, W2T, 1024, 256);

    dim3 blk(128);
    const int mt = MROWS / BM;   // 1250

    // 2) value = src @ W_val + b_val  -> bf16
    gemm_ker<0><<<dim3(2, mt), blk, 0, stream>>>(src, WvT, b_val, nullptr, valB,
                                                 MROWS, 256, 256, 0);
    // 3) sampling offsets / attention logits
    gemm_ker<0><<<dim3(1, mt), blk, 0, stream>>>(query, WoffT, b_off, offF, nullptr,
                                                 MROWS, 64, 256, 0);
    gemm_ker<0><<<dim3(1, mt), blk, 0, stream>>>(query, WawT, b_aw, awF, nullptr,
                                                 MROWS, 32, 256, 0);
    // 4) softmax + bilinear gather + weighted sum  -> attn (bf16)
    msda_sample<<<MROWS, 256, 0, stream>>>(offF, awF, refp, valB, attnB);
    // 5) attn @ W_out + b_out -> fp32
    gemm_ker<1><<<dim3(2, mt), blk, 0, stream>>>(attnB, WoutT, b_out, scrF, nullptr,
                                                 MROWS, 256, 256, 0);
    // 6) q = LN(query + attn_out)  (fp32 for residual, bf16 for next GEMM)
    add_ln<<<MROWS / 8, 256, 0, stream>>>(query, scrF, g1, be1, qF, qB, MROWS);
    // 7) h1 = relu(q @ W1 + b1) -> bf16
    gemm_ker<1><<<dim3(8, mt), blk, 0, stream>>>(qB, W1T, b1, nullptr, h1,
                                                 MROWS, 1024, 256, 1);
    // 8) ffn = h1 @ W2 + b2 -> fp32 (reuse scrF)
    gemm_ker<1><<<dim3(2, mt), blk, 0, stream>>>(h1, W2T, b2, scrF, nullptr,
                                                 MROWS, 256, 1024, 0);
    // 9) out = LN(q + ffn)
    add_ln<<<MROWS / 8, 256, 0, stream>>>(qF, scrF, g2, be2, out, nullptr, MROWS);
}
